// Q_network_74302934220962
// MI455X (gfx1250) — matrix operand order repaired and checked by reading, not compiled, not run on hardware
//
#include <hip/hip_runtime.h>
#include <math.h>
#include <stdint.h>

typedef __attribute__((ext_vector_type(16))) _Float16 v16h;
typedef __attribute__((ext_vector_type(8)))  float    v8f;

#define B_   128
#define T_   64
#define A_   27
#define E_   30
#define D_   64
#define BT_  (B_ * T_)            // 8192
#define RA_  (BT_ * A_)           // 221184 ally rows
#define RE_  (BT_ * E_)           // 245760 enemy rows
#define RBA_ (B_ * A_)            // 3456 GRU sequences

__device__ __forceinline__ float gelu_erf(float x) {
    return 0.5f * x * (1.0f + erff(x * 0.7071067811865475f));
}

// Async copy of 16 bytes global -> LDS (CDNA5 GLOBAL_LOAD_ASYNC_TO_LDS_B128,
// tracked by ASYNCcnt). ldsaddr = LDS byte address (low 32 bits of generic
// pointer to __shared__), voff = byte offset added to uniform saddr base.
__device__ __forceinline__ void async_copy_b128(uint32_t ldsaddr, uint32_t voff,
                                                const float* base) {
    asm volatile("global_load_async_to_lds_b128 %0, %1, %2"
                 :: "v"(ldsaddr), "v"(voff), "s"(base) : "memory");
}
__device__ __forceinline__ void async_wait0() {
    asm volatile("s_wait_asynccnt 0" ::: "memory");
}

// ---------------------------------------------------------------------------
// Generic fused GEMM: Y[R,64] = epi( concat(X1,X2)[R,Ktot] @ W[64,Ktot]^T + b )
// Block = 128 threads (4 waves). Each block: 16 rows. Wave w: cols [16w,16w+16).
// K staged to LDS (f32) in 32-chunks; ASYNC path uses global_load_async_to_lds
// (requires K1%32==0 and no padding); WMMA f32_16x16x32_f16 accumulate.
// Epilogue: bias (+GELU) (+residual) (+LayerNorm over the 64 outputs).
// ---------------------------------------------------------------------------
template<bool GELU, bool LN, bool RESID, bool ASYNC>
__global__ __launch_bounds__(128)
void gemm64(const float* __restrict__ X1, int K1,
            const float* __restrict__ X2, int K2,
            const float* __restrict__ W,
            const float* __restrict__ bias,
            const float* __restrict__ gamma,
            const float* __restrict__ beta,
            const float* __restrict__ resid,
            float* __restrict__ Y)
{
    const int Ktot   = K1 + K2;
    const int kTiles = (Ktot + 31) >> 5;
    const int m0     = blockIdx.x << 4;
    const int tid    = threadIdx.x;
    const int lane   = tid & 31;
    const int wave   = tid >> 5;

    __shared__ float laf[16][32];
    __shared__ float ly[16][64];
    __shared__ float lmv[16][2];

    v8f acc = {};
    for (int kt = 0; kt < kTiles; ++kt) {
        if (ASYNC) {
            // full aligned tile; source (and stride) uniform per k-tile
            const float* src; int stride, kofs;
            if ((kt << 5) < K1) { src = X1; stride = K1; kofs = (kt << 5); }
            else                { src = X2; stride = K2; kofs = (kt << 5) - K1; }
            const int r  = tid >> 3;
            const int c4 = (tid & 7) << 2;
            uint32_t voff = (uint32_t)(((m0 + r) * stride + kofs + c4) * 4);
            uint32_t lds  = (uint32_t)(uintptr_t)&laf[r][c4];
            async_copy_b128(lds, voff, src);
            async_wait0();
        } else {
            for (int idx = tid; idx < 16 * 32; idx += 128) {
                int r = idx >> 5, c = idx & 31;
                int kg = (kt << 5) + c;
                float v = 0.f;
                if (kg < K1)        v = X1[(size_t)(m0 + r) * K1 + kg];
                else if (kg < Ktot) v = X2[(size_t)(m0 + r) * K2 + (kg - K1)];
                laf[r][c] = v;
            }
        }
        __syncthreads();
        // A fragment: lanes 0-15: M=lane, K 0..15; lanes 16-31: K 16..31
        v16h a;
        {
            int m  = lane & 15;
            int kb = (lane >> 4) << 4;
            #pragma unroll
            for (int e = 0; e < 8; ++e) {
                a[e]     = (_Float16)laf[m][kb + e];
                a[8 + e] = (_Float16)laf[m][8 + kb + e];
            }
        }
        // B fragment: B[k][n] = W[n][k]; lane n = 16w + lane&15; lanes<16: K 0..15, else 16..31
        v16h b;
        {
            int n  = (wave << 4) + (lane & 15);
            int kb = (lane >> 4) << 4;
            const float* wr = W + (size_t)n * Ktot + (kt << 5) + kb;
            #pragma unroll
            for (int e = 0; e < 16; ++e) {
                int kg = (kt << 5) + kb + e;
                b[e] = (kg < Ktot) ? (_Float16)wr[e] : (_Float16)0.f;
            }
        }
        acc = __builtin_amdgcn_wmma_f32_16x16x32_f16(false, a, false, b, (short)0, acc, false, false);
        __syncthreads();
    }

    // epilogue: C/D layout: VGPR r -> row (lane<16 ? r : r+8), col = 16w + lane&15
    const int col   = (wave << 4) + (lane & 15);
    const int rbase = (lane >> 4) << 3;
    const float bv  = bias[col];
    #pragma unroll
    for (int r = 0; r < 8; ++r) {
        int row  = rbase + r;
        float v  = acc[r] + bv;
        if (GELU)  v = gelu_erf(v);
        if (RESID) v += resid[(size_t)(m0 + row) * 64 + col];
        if (LN) ly[row][col] = v;
        else    Y[(size_t)(m0 + row) * 64 + col] = v;
    }
    if (LN) {
        __syncthreads();
        if (tid < 16) {
            float s = 0.f;
            #pragma unroll
            for (int c = 0; c < 64; ++c) s += ly[tid][c];
            float m = s * (1.f / 64.f);
            float vv = 0.f;
            #pragma unroll
            for (int c = 0; c < 64; ++c) { float d = ly[tid][c] - m; vv += d * d; }
            lmv[tid][0] = m;
            lmv[tid][1] = rsqrtf(vv * (1.f / 64.f) + 1e-5f);
        }
        __syncthreads();
        for (int idx = tid; idx < 16 * 64; idx += 128) {
            int r = idx >> 6, c = idx & 63;
            Y[(size_t)(m0 + r) * 64 + c] =
                (ly[r][c] - lmv[r][0]) * lmv[r][1] * gamma[c] + beta[c];
        }
    }
}

// ---------------------------------------------------------------------------
// Per-sample multi-head attention (H=4, dh=16), queries over A_=27 rows,
// keys/values over NKV rows. DIAG: mask diagonal + agent mask, force fw[i][i]=1.
// Output: LayerNorm(residual + aggregate).
// ---------------------------------------------------------------------------
template<int NKV, bool DIAG>
__global__ __launch_bounds__(128)
void attn_kernel(const float* __restrict__ Q,   // [BT, A_, 64]
                 const float* __restrict__ Kv,  // [BT, NKV, 64]
                 const float* __restrict__ Vv,  // [BT, NKV, 64]
                 const float* __restrict__ RES, // [BT, A_, 64]
                 const float* __restrict__ am,  // [BT, A_] (DIAG only)
                 const float* __restrict__ gamma,
                 const float* __restrict__ beta,
                 float* __restrict__ Y)         // [BT, A_, 64]
{
    const int bt  = blockIdx.x;
    const int tid = threadIdx.x;
    __shared__ float kb[NKV][64];
    __shared__ float vb[NKV][64];
    __shared__ float ab[A_][64];
    __shared__ float mrow[NKV > A_ ? NKV : A_];
    __shared__ float mv[A_][2];

    const size_t qbase = (size_t)bt * A_ * 64;
    const size_t kbase = (size_t)bt * NKV * 64;
    for (int idx = tid; idx < NKV * 64; idx += 128) {
        kb[idx / 64][idx & 63] = Kv[kbase + idx];
        vb[idx / 64][idx & 63] = Vv[kbase + idx];
    }
    if (DIAG) { if (tid < A_) mrow[tid] = am[(size_t)bt * A_ + tid]; }
    __syncthreads();

    if (tid < 4 * A_) {
        const int h = tid / A_, i = tid % A_;
        float qr[16];
        const float* qp = Q + qbase + (size_t)i * 64 + h * 16;
        #pragma unroll
        for (int d = 0; d < 16; ++d) qr[d] = qp[d];
        float sc[NKV];
        float mx = -3.4e38f;
        for (int j = 0; j < NKV; ++j) {
            float s = 0.f;
            #pragma unroll
            for (int d = 0; d < 16; ++d) s += qr[d] * kb[j][h * 16 + d];
            s *= 0.25f;                       // 1/sqrt(dh), dh=16
            if (DIAG) { if (j == i || mrow[j] == 0.f) s = -3.4e38f; }
            sc[j] = s;
            mx = fmaxf(mx, s);
        }
        float sum = 0.f;
        for (int j = 0; j < NKV; ++j) { sc[j] = expf(sc[j] - mx); sum += sc[j]; }
        const float inv = 1.f / sum;
        #pragma unroll
        for (int d = 0; d < 16; ++d) {
            float a = 0.f;
            for (int j = 0; j < NKV; ++j) {
                float fw = sc[j] * inv;
                if (DIAG) { if (j == i) fw = 1.0f; }
                a += fw * vb[j][h * 16 + d];
            }
            ab[i][h * 16 + d] = a;
        }
    }
    __syncthreads();
    if (tid < A_) {
        float s = 0.f;
        for (int c = 0; c < 64; ++c) {
            float v = ab[tid][c] + RES[qbase + (size_t)tid * 64 + c];
            ab[tid][c] = v; s += v;
        }
        float m = s * (1.f / 64.f);
        float vv = 0.f;
        for (int c = 0; c < 64; ++c) { float d = ab[tid][c] - m; vv += d * d; }
        mv[tid][0] = m;
        mv[tid][1] = rsqrtf(vv * (1.f / 64.f) + 1e-5f);
    }
    __syncthreads();
    for (int idx = tid; idx < A_ * 64; idx += 128) {
        int r = idx / 64, c = idx & 63;
        Y[qbase + idx] = (ab[r][c] - mv[r][0]) * mv[r][1] * gamma[c] + beta[c];
    }
}

// ---------------------------------------------------------------------------
// Persistent GRU: each block owns 16 of the B*A=3456 sequences, loops T=64
// steps. Per step: x tile async-staged to LDS; A-fragments (x and h) built
// once; gi = x@Wih^T + bih and gh = h@Whh^T + bhh via WMMA (12 n-tiles of 16
// over 192 gate cols, K=64 as two 32-chunks), then gate update.
// Output written directly in [B,T,A,D] order.
// ---------------------------------------------------------------------------
__global__ __launch_bounds__(128)
void gru_kernel(const float* __restrict__ feat,  // [(b*T+t)*A + a, 64]
                const float* __restrict__ h0,    // [B*A, 64]
                const float* __restrict__ Wih,   // [192, 64]
                const float* __restrict__ Whh,   // [192, 64]
                const float* __restrict__ bih,
                const float* __restrict__ bhh,
                float* __restrict__ Yout)        // [B,T,A,64]
{
    const int r0   = blockIdx.x << 4;
    const int tid  = threadIdx.x;
    const int lane = tid & 31;
    const int wave = tid >> 5;

    __shared__ _Float16 hx[16][64];
    __shared__ float    hf[16][64];
    __shared__ float    xbf[16][64];
    __shared__ float    gi[16][192];
    __shared__ float    gh[16][192];

    for (int idx = tid; idx < 16 * 64; idx += 128) {
        int r = idx >> 6, c = idx & 63;
        float v = h0[(size_t)(r0 + r) * 64 + c];
        hf[r][c] = v;
        hx[r][c] = (_Float16)v;
    }
    __syncthreads();

    for (int t = 0; t < T_; ++t) {
        // async-stage x tile: 16 rows x 64 f32 = 256 x 16B chunks
        for (int cid = tid; cid < 256; cid += 128) {
            int r = cid >> 4, c4 = (cid & 15) << 2;
            int gr = r0 + r;
            int bidx = gr / A_, a = gr - bidx * A_;
            uint32_t voff = (uint32_t)(((((bidx * T_) + t) * A_ + a) * 64 + c4) * 4);
            uint32_t lds  = (uint32_t)(uintptr_t)&xbf[r][c4];
            async_copy_b128(lds, voff, feat);
        }
        async_wait0();
        __syncthreads();

        // build A fragments once (independent of n-tile)
        v16h afx[2], afh[2];
        {
            const int m = lane & 15;
            #pragma unroll
            for (int ks = 0; ks < 2; ++ks) {
                const int kb = ((lane >> 4) << 4) + ks * 32;
                #pragma unroll
                for (int e = 0; e < 8; ++e) {
                    afx[ks][e]     = (_Float16)xbf[m][kb + e];
                    afx[ks][8 + e] = (_Float16)xbf[m][kb + 8 + e];
                    afh[ks][e]     = hx[m][kb + e];
                    afh[ks][8 + e] = hx[m][kb + 8 + e];
                }
            }
        }

        for (int nt = wave; nt < 12; nt += 4) {
            v8f ai = {}, ah = {};
            const int n = (nt << 4) + (lane & 15);
            #pragma unroll
            for (int ks = 0; ks < 2; ++ks) {
                const int kw = ((lane >> 4) << 4) + ks * 32;
                v16h bf;
                const float* wr = Wih + (size_t)n * 64 + kw;
                #pragma unroll
                for (int e = 0; e < 16; ++e) bf[e] = (_Float16)wr[e];
                ai = __builtin_amdgcn_wmma_f32_16x16x32_f16(false, afx[ks], false, bf, (short)0, ai, false, false);
                const float* wr2 = Whh + (size_t)n * 64 + kw;
                #pragma unroll
                for (int e = 0; e < 16; ++e) bf[e] = (_Float16)wr2[e];
                ah = __builtin_amdgcn_wmma_f32_16x16x32_f16(false, afh[ks], false, bf, (short)0, ah, false, false);
            }
            const int col = (nt << 4) + (lane & 15);
            const int rb  = (lane >> 4) << 3;
            const float bi = bih[col], bh = bhh[col];
            #pragma unroll
            for (int r = 0; r < 8; ++r) {
                gi[rb + r][col] = ai[r] + bi;
                gh[rb + r][col] = ah[r] + bh;
            }
        }
        __syncthreads();

        for (int idx = tid; idx < 16 * 64; idx += 128) {
            int r = idx >> 6, d = idx & 63;
            float ir = gi[r][d],       hr = gh[r][d];
            float iz = gi[r][64 + d],  hz = gh[r][64 + d];
            float in = gi[r][128 + d], hn = gh[r][128 + d];
            float rg = 1.f / (1.f + expf(-(ir + hr)));
            float zg = 1.f / (1.f + expf(-(iz + hz)));
            float ng = tanhf(in + rg * hn);
            float hnew = (1.f - zg) * ng + zg * hf[r][d];
            hf[r][d] = hnew;
            hx[r][d] = (_Float16)hnew;
            int gr = r0 + r;
            int bidx = gr / A_, a = gr - bidx * A_;
            size_t orow = ((size_t)bidx * T_ + t) * A_ + a;
            Yout[orow * 64 + d] = hnew;
        }
        __syncthreads();
    }
}

// ---------------------------------------------------------------------------
// Final head: qv = LN(x) @ qv_W^T + qv_b (N=1)
// ---------------------------------------------------------------------------
__global__ __launch_bounds__(256)
void qv_kernel(const float* __restrict__ X, const float* __restrict__ g,
               const float* __restrict__ b, const float* __restrict__ qw,
               const float* __restrict__ qb, float* __restrict__ out, int R)
{
    int r = blockIdx.x * blockDim.x + threadIdx.x;
    if (r >= R) return;
    const float* x = X + (size_t)r * 64;
    float s = 0.f;
    #pragma unroll
    for (int c = 0; c < 64; ++c) s += x[c];
    float m = s * (1.f / 64.f), vv = 0.f;
    #pragma unroll
    for (int c = 0; c < 64; ++c) { float d = x[c] - m; vv += d * d; }
    float inv = rsqrtf(vv * (1.f / 64.f) + 1e-5f);
    float acc = qb[0];
    #pragma unroll
    for (int c = 0; c < 64; ++c)
        acc += ((x[c] - m) * inv * g[c] + b[c]) * qw[c];
    out[r] = acc;
}

// ---------------------------------------------------------------------------
extern "C" void kernel_launch(void* const* d_in, const int* in_sizes, int n_in,
                              void* d_out, int out_size, void* d_ws, size_t ws_size,
                              hipStream_t stream)
{
    (void)in_sizes; (void)n_in; (void)out_size; (void)ws_size;
    const float* states  = (const float*)d_in[0];   // [B,T,A,96]
    const float* estates = (const float*)d_in[1];   // [B,T,E,64]
    const float* actions = (const float*)d_in[2];   // [B,T,A,36]
    const float* h0      = (const float*)d_in[3];   // [1,B*A,64]
    const float* amask   = (const float*)d_in[4];   // [B,T,A]
    const int P = 5;
    #define PAR(i) ((const float*)d_in[P + (i)])
    const float *ally_W = PAR(0),  *ally_b = PAR(1),  *ally_g = PAR(2),  *ally_be = PAR(3);
    const float *en_W   = PAR(4),  *en_b   = PAR(5),  *en_g   = PAR(6),  *en_be   = PAR(7);
    const float *act_W  = PAR(8),  *act_b  = PAR(9),  *act_g  = PAR(10), *act_be  = PAR(11);
    const float *k_W = PAR(12), *k_b = PAR(13), *q_W = PAR(14), *q_b = PAR(15);
    const float *v_W = PAR(16), *v_b = PAR(17);
    const float *attn_g = PAR(18), *attn_be = PAR(19);
    const float *ff1_W = PAR(20), *ff1_b = PAR(21), *ff_g = PAR(22), *ff_be = PAR(23);
    const float *ff2_W = PAR(24), *ff2_b = PAR(25), *ffo_g = PAR(26), *ffo_be = PAR(27);
    const float *ke_W = PAR(28), *ke_b = PAR(29), *qe_W = PAR(30), *qe_b = PAR(31);
    const float *ve_W = PAR(32), *ve_b = PAR(33);
    const float *atne_g = PAR(34), *atne_be = PAR(35);
    const float *fe1_W = PAR(36), *fe1_b = PAR(37), *fe_g = PAR(38), *fe_be = PAR(39);
    const float *fe2_W = PAR(40), *fe2_b = PAR(41), *feo_g = PAR(42), *feo_be = PAR(43);
    const float *cm_W = PAR(44), *cm_b = PAR(45);
    const float *Wih = PAR(46), *Whh = PAR(47), *bih = PAR(48), *bhh = PAR(49);
    const float *qln_g = PAR(50), *qln_b = PAR(51), *qv_W = PAR(52), *qv_b = PAR(53);
    #undef PAR

    float* ws = (float*)d_ws;
    const size_t SA = (size_t)RA_ * 64;   // ally slot
    const size_t SE = (size_t)RE_ * 64;   // enemy slot
    float* S0 = ws + 0 * SA;   // se -> later enemy_out
    float* S2 = ws + 1 * SA;   // oae -> later feat
    float* S3 = ws + 2 * SA;   // q -> ff1out -> qe -> gru out
    float* S4 = ws + 3 * SA;   // k -> agge_
    float* S5 = ws + 4 * SA;   // v -> ally_out
    float* S6 = ws + 5 * SA;   // agg_ -> ffe1out
    float* E0 = ws + 6 * SA;            // ee
    float* E1 = E0 + SE;                // ke
    float* E2 = E1 + SE;                // ve

    const int GA = RA_ / 16;   // 13824
    const int GE = RE_ / 16;   // 15360

    // embeddings
    gemm64<true, true, false, true><<<GA, 128, 0, stream>>>(states, 96, nullptr, 0,
        ally_W, ally_b, ally_g, ally_be, nullptr, S0);                       // se
    gemm64<true, true, false, true><<<GE, 128, 0, stream>>>(estates, 64, nullptr, 0,
        en_W, en_b, en_g, en_be, nullptr, E0);                               // ee
    gemm64<true, true, false, false><<<GA, 128, 0, stream>>>(states, 96, actions, 36,
        act_W, act_b, act_g, act_be, nullptr, S2);                           // oae (padded K)

    // ally self-attention
    gemm64<false, false, false, true><<<GA, 128, 0, stream>>>(S0, 64, nullptr, 0,
        q_W, q_b, nullptr, nullptr, nullptr, S3);                            // q
    gemm64<false, false, false, true><<<GA, 128, 0, stream>>>(S0, 64, nullptr, 0,
        k_W, k_b, nullptr, nullptr, nullptr, S4);                            // k
    gemm64<false, false, false, true><<<GA, 128, 0, stream>>>(S2, 64, nullptr, 0,
        v_W, v_b, nullptr, nullptr, nullptr, S5);                            // v
    attn_kernel<A_, true><<<BT_, 128, 0, stream>>>(S3, S4, S5, S2, amask,
        attn_g, attn_be, S6);                                                // agg_
    gemm64<true, true, false, true><<<GA, 128, 0, stream>>>(S6, 64, nullptr, 0,
        ff1_W, ff1_b, ff_g, ff_be, nullptr, S3);                             // ff mid
    gemm64<false, true, true, true><<<GA, 128, 0, stream>>>(S3, 64, nullptr, 0,
        ff2_W, ff2_b, ffo_g, ffo_be, S6, S5);                                // ally_out

    // agent->enemy cross-attention
    gemm64<false, false, false, true><<<GA, 128, 0, stream>>>(S0, 64, nullptr, 0,
        qe_W, qe_b, nullptr, nullptr, nullptr, S3);                          // qe
    gemm64<false, false, false, true><<<GE, 128, 0, stream>>>(E0, 64, nullptr, 0,
        ke_W, ke_b, nullptr, nullptr, nullptr, E1);                          // ke
    gemm64<false, false, false, true><<<GE, 128, 0, stream>>>(E0, 64, nullptr, 0,
        ve_W, ve_b, nullptr, nullptr, nullptr, E2);                          // ve
    attn_kernel<E_, false><<<BT_, 128, 0, stream>>>(S3, E1, E2, S0, nullptr,
        atne_g, atne_be, S4);                                                // agge_
    gemm64<true, true, false, true><<<GA, 128, 0, stream>>>(S4, 64, nullptr, 0,
        fe1_W, fe1_b, fe_g, fe_be, nullptr, S6);                             // ffe mid
    gemm64<false, true, true, true><<<GA, 128, 0, stream>>>(S6, 64, nullptr, 0,
        fe2_W, fe2_b, feo_g, feo_be, S4, S0);                                // enemy_out

    // common features (concat of two aligned 64-wide sources)
    gemm64<true, false, false, true><<<GA, 128, 0, stream>>>(S5, 64, S0, 64,
        cm_W, cm_b, nullptr, nullptr, nullptr, S2);                          // feat

    // GRU over T, output in [B,T,A,D]
    gru_kernel<<<RBA_ / 16, 128, 0, stream>>>(S2, h0, Wih, Whh, bih, bhh, S3);

    // final head
    qv_kernel<<<(RA_ + 255) / 256, 256, 0, stream>>>(S3, qln_g, qln_b, qv_W, qv_b,
        (float*)d_out, RA_);
}
